// ResBlock_1_77137612636471
// MI455X (gfx1250) — 8-bit matrix operand order repaired and checked by reading, compiled, not run on hardware
//
#include <hip/hip_runtime.h>

typedef int v8i __attribute__((ext_vector_type(8)));

#define C_CH   256
#define HWSZ   3136
#define NB     32
#define NPIX   (NB*HWSZ)
#define HP     58
#define K_TOT  2304
#define EPSV   1e-5f

#define OFF_Y16   ((size_t)0)
#define OFF_XB    ((size_t)51380224)
#define OFF_WB    (OFF_XB + (size_t)27557888)
#define OFF_SUM   (OFF_WB + (size_t)589824)
#define OFF_SQ    (OFF_SUM + 1024)
#define OFF_SC    (OFF_SQ  + 1024)
#define OFF_SH    (OFF_SC  + 1024)

__device__ __forceinline__ signed char sgn8(float v) {
    return (signed char)((v > 0.f) ? 1 : ((v < 0.f) ? -1 : 0));
}

#define XPAD 264
__global__ void __launch_bounds__(256)
pack_x_kernel(const float* __restrict__ x, signed char* __restrict__ xb) {
    __shared__ signed char lds[56 * XPAD];
    const int hp = blockIdx.x % HP;
    const int n  = blockIdx.x / HP;
    int* xbrow = (int*)xb + ((long)(n * HP + hp) * HP) * 64;
    const int t = threadIdx.x;

    if (hp >= 1 && hp <= 56) {
        const int h = hp - 1;
        #pragma unroll 4
        for (int i = 0; i < 56; ++i) {
            int e = i * 256 + t;
            int c = e / 56;
            int w = e % 56;
            float v = x[((long)n * C_CH + c) * HWSZ + h * 56 + w];
            lds[w * XPAD + c] = sgn8(v);
        }
        __syncthreads();
        #pragma unroll
        for (int i = 0; i < 14; ++i) {
            int e  = i * 256 + t;
            int w  = e >> 6;
            int c4 = e & 63;
            int val = *(const int*)(lds + w * XPAD + c4 * 4);
            xbrow[(w + 1) * 64 + c4] = val;
        }
        if (t < 128) {
            int wp = (t >> 6) ? 57 : 0;
            xbrow[wp * 64 + (t & 63)] = 0;
        }
    } else {
        for (int i = t; i < HP * 64; i += 256) xbrow[i] = 0;
    }
}

__global__ void pack_w_kernel(const float* __restrict__ wt, signed char* __restrict__ wb) {
    int idx = blockIdx.x * blockDim.x + threadIdx.x;
    if (idx >= C_CH * 9 * 64) return;
    int c4 = idx & 63;
    int kpos = (idx >> 6) % 9;
    int o = idx / (64 * 9);
    int pack = 0;
    #pragma unroll
    for (int j = 0; j < 4; ++j) {
        int ci = c4 * 4 + j;
        signed char b = sgn8(wt[(long)o * K_TOT + ci * 9 + kpos]);
        pack |= ((int)(unsigned char)b) << (8 * j);
    }
    ((int*)wb)[o * (K_TOT / 4) + kpos * 64 + c4] = pack;
}

__global__ void zero_stats_kernel(float* __restrict__ sum, float* __restrict__ sq) {
    int t = threadIdx.x;
    if (t < C_CH) { sum[t] = 0.f; sq[t] = 0.f; }
}

__global__ void __launch_bounds__(256)
conv_wmma_kernel(const signed char* __restrict__ xb,
                 const signed char* __restrict__ wb,
                 short* __restrict__ y16,
                 float* __restrict__ sum, float* __restrict__ sq) {
    const int wid  = threadIdx.x >> 5;
    const int lane = threadIdx.x & 31;
    const int half = lane >> 4;
    const int lm   = lane & 15;

    const int pbase = blockIdx.x * 64  + (wid & 1) * 32;
    const int cbase = blockIdx.y * 128 + (wid >> 1) * 32;

    const int p0 = pbase + lm,        p1 = p0 + 16;
    const int n0 = p0 / HWSZ, hw0 = p0 % HWSZ, h0 = hw0 / 56, w0 = hw0 % 56;
    const int n1 = p1 / HWSZ, hw1 = p1 % HWSZ, h1 = hw1 / 56, w1 = hw1 % 56;

    const int o0 = cbase + lm, o1 = o0 + 16;
    const int* wcol0 = (const int*)wb + o0 * (K_TOT / 4);
    const int* wcol1 = (const int*)wb + o1 * (K_TOT / 4);

    v8i acc00 = {0,0,0,0,0,0,0,0};
    v8i acc01 = {0,0,0,0,0,0,0,0};
    v8i acc10 = {0,0,0,0,0,0,0,0};
    v8i acc11 = {0,0,0,0,0,0,0,0};

    #pragma unroll
    for (int s = 0; s < 36; ++s) {
        const int kpos = s >> 2;
        const int cb   = (s & 3) * 64;
        const int dh   = kpos / 3;
        const int dw   = kpos % 3;

        const int* ap0 = (const int*)xb +
            ((((n0 * HP) + h0 + dh) * HP + (w0 + dw)) << 6) + (cb >> 2);
        const int* ap1 = (const int*)xb +
            ((((n1 * HP) + h1 + dh) * HP + (w1 + dw)) << 6) + (cb >> 2);
        const int* bp0 = wcol0 + s * 16;
        const int* bp1 = wcol1 + s * 16;

        v8i A0, A1, B0, B1;
        #pragma unroll
        for (int v = 0; v < 8; ++v) {
            const int ka = ((v >= 4) ? 32 : 0) + (((v >> 1) & 1) << 4) + ((v & 1) << 2) + (half << 3);
            const int kb = ((v >= 4) ? 32 : 0) + (((v >> 1) & 1) << 4) + ((v & 1) << 2) + (half << 3);
            A0[v] = ap0[ka >> 2];
            A1[v] = ap1[ka >> 2];
            B0[v] = bp0[kb >> 2];
            B1[v] = bp1[kb >> 2];
        }
        acc00 = __builtin_amdgcn_wmma_i32_16x16x64_iu8(true, A0, true, B0, acc00, false, false);
        acc01 = __builtin_amdgcn_wmma_i32_16x16x64_iu8(true, A0, true, B1, acc01, false, false);
        acc10 = __builtin_amdgcn_wmma_i32_16x16x64_iu8(true, A1, true, B0, acc10, false, false);
        acc11 = __builtin_amdgcn_wmma_i32_16x16x64_iu8(true, A1, true, B1, acc11, false, false);
    }

    float s0 = 0.f, q0 = 0.f, s1 = 0.f, q1 = 0.f;
    #pragma unroll
    for (int r = 0; r < 8; ++r) {
        const int m   = r + (half << 3);
        const int pr0 = pbase + m;
        const int pr1 = pbase + 16 + m;
        const int v00 = acc00[r], v01 = acc01[r], v10 = acc10[r], v11 = acc11[r];
        y16[(long)pr0 * C_CH + o0] = (short)v00;
        y16[(long)pr0 * C_CH + o1] = (short)v01;
        y16[(long)pr1 * C_CH + o0] = (short)v10;
        y16[(long)pr1 * C_CH + o1] = (short)v11;
        float f00 = (float)v00, f01 = (float)v01, f10 = (float)v10, f11 = (float)v11;
        s0 += f00 + f10;  q0 += f00 * f00 + f10 * f10;
        s1 += f01 + f11;  q1 += f01 * f01 + f11 * f11;
    }
    s0 += __shfl_xor(s0, 16, 32);
    q0 += __shfl_xor(q0, 16, 32);
    s1 += __shfl_xor(s1, 16, 32);
    q1 += __shfl_xor(q1, 16, 32);
    if (half == 0) {
        atomicAdd(&sum[o0], s0);
        atomicAdd(&sq[o0],  q0);
        atomicAdd(&sum[o1], s1);
        atomicAdd(&sq[o1],  q1);
    }
}

__global__ void stats_kernel(const float* __restrict__ sum, const float* __restrict__ sq,
                             const float* __restrict__ gamma, const float* __restrict__ beta,
                             float* __restrict__ scale, float* __restrict__ shift) {
    int c = threadIdx.x;
    if (c >= C_CH) return;
    const float inv_n = 1.f / (float)NPIX;
    float mean = sum[c] * inv_n;
    float var  = sq[c] * inv_n - mean * mean;
    float istd = rsqrtf(var + EPSV);
    float sc   = gamma[c] * istd;
    scale[c] = sc;
    shift[c] = beta[c] - mean * sc;
}

#define YST 67
__global__ void __launch_bounds__(256)
epilogue_kernel(const float* __restrict__ x,
                const short* __restrict__ y16,
                const float* __restrict__ scale,
                const float* __restrict__ shift,
                float* __restrict__ out) {
    __shared__ short lds[C_CH * YST];
    const int pb  = blockIdx.x * 64;
    const int n   = pb / HWSZ;
    const int hwb = pb % HWSZ;
    const int t   = threadIdx.x;

    #pragma unroll 4
    for (int i = 0; i < 64; i += 2) {
        int pr = i + (t >> 7);
        int c2 = t & 127;
        int v  = ((const int*)y16)[(long)(pb + pr) * (C_CH / 2) + c2];
        lds[(2 * c2 + 0) * YST + pr] = (short)(v & 0xFFFF);
        lds[(2 * c2 + 1) * YST + pr] = (short)((unsigned)v >> 16);
    }
    __syncthreads();

    const int p  = t & 63;
    const int c0 = t >> 6;
    #pragma unroll 4
    for (int cc = c0; cc < C_CH; cc += 4) {
        long oidx = ((long)n * C_CH + cc) * HWSZ + hwb + p;
        float y = (float)lds[cc * YST + p];
        out[oidx] = x[oidx] + y * scale[cc] + shift[cc];
    }
}

extern "C" void kernel_launch(void* const* d_in, const int* in_sizes, int n_in,
                              void* d_out, int out_size, void* d_ws, size_t ws_size,
                              hipStream_t stream) {
    const float* x     = (const float*)d_in[0];
    const float* wt    = (const float*)d_in[1];
    const float* gamma = (const float*)d_in[2];
    const float* beta  = (const float*)d_in[3];
    float* out = (float*)d_out;

    char* ws = (char*)d_ws;
    short*       y16 = (short*)(ws + OFF_Y16);
    signed char* xb  = (signed char*)(ws + OFF_XB);
    signed char* wb  = (signed char*)(ws + OFF_WB);
    float*       sum = (float*)(ws + OFF_SUM);
    float*       sq  = (float*)(ws + OFF_SQ);
    float*       sc  = (float*)(ws + OFF_SC);
    float*       sh  = (float*)(ws + OFF_SH);

    pack_x_kernel<<<NB * HP, 256, 0, stream>>>(x, xb);
    {
        int total = C_CH * 9 * 64;
        pack_w_kernel<<<(total + 255) / 256, 256, 0, stream>>>(wt, wb);
    }
    zero_stats_kernel<<<1, 256, 0, stream>>>(sum, sq);
    {
        dim3 grid(NPIX / 64, C_CH / 128);
        conv_wmma_kernel<<<grid, 256, 0, stream>>>(xb, wb, y16, sum, sq);
    }
    stats_kernel<<<1, 256, 0, stream>>>(sum, sq, gamma, beta, sc, sh);
    epilogue_kernel<<<NPIX / 64, 256, 0, stream>>>(x, y16, sc, sh, out);
}
